// CIELUVChannelLoss_50483045597470
// MI455X (gfx1250) — compile-verified, matrix operand order checked by reading, not run on hardware
//
#include <hip/hip_runtime.h>

typedef __attribute__((ext_vector_type(16))) _Float16 v16h;
typedef __attribute__((ext_vector_type(8)))  float    v8f;

#define HW      512
#define NPLANE  (512 * 512)
#define NIMG    (3 * 512 * 512)
#define NPIX    (16 * 512 * 512)        // per-channel element count
#define NTILES  16384                   // 16 images * 32 * 32 tiles

// ---------------------------------------------------------------------------
// Fast cube root for t in (0.008856, ~1.05]: hw log2/exp2 + one Newton step.
// v_log_f32 / v_exp_f32 are base-2 on AMDGPU.
// ---------------------------------------------------------------------------
__device__ __forceinline__ float cbrt_pos(float t) {
    float c = __builtin_amdgcn_exp2f(0.33333333333333333f * __builtin_amdgcn_logf(t));
    float r = __builtin_amdgcn_rcpf(c * c);
    return (2.0f * c + t * r) * 0.33333334f;
}

__device__ __forceinline__ float fcie(float t) {
    // reference: t > 0.008856 ? cbrt(t) : 7.787*t + 16/116
    return t > 0.008856f ? cbrt_pos(t) : 7.787f * t + (16.0f / 116.0f);
}

__device__ __forceinline__ void rgb2luv(float r, float g, float b,
                                        float& l, float& u, float& v) {
    float x = 0.4124564f * r + 0.3575761f * g + 0.1804375f * b;
    float y = 0.2126729f * r + 0.7151522f * g + 0.0721750f * b;
    float z = 0.0193339f * r + 0.1191920f * g + 0.9503041f * b;
    x *= (1.0f / 0.95047f);
    z *= (1.0f / 1.08883f);
    float fx = fcie(x), fy = fcie(y), fz = fcie(z);
    l = 116.0f * fy - 16.0f;
    u = 13.0f * l * (fx - fy);
    v = 13.0f * l * (fy - fz);
}

// ---------------------------------------------------------------------------
// Kernel 1 (path A): pointwise LUV diff, each pixel converted exactly once.
// D is stored NCHW (same layout as the inputs' planes).
// ---------------------------------------------------------------------------
__global__ __launch_bounds__(256) void luv_diff_kernel(
        const float* __restrict__ inp, const float* __restrict__ tgt,
        float* __restrict__ D) {
    const int idx = blockIdx.x * 256 + threadIdx.x;       // pixel id in [0, NPIX)
    const int n = idx >> 18;                              // / 262144
    const size_t off = (size_t)idx + (size_t)n * (2 * NPLANE);  // (n,0,h,w)
    float l0, u0, v0, l1, u1, v1;
    rgb2luv(inp[off], inp[off + NPLANE], inp[off + 2 * NPLANE], l0, u0, v0);
    rgb2luv(tgt[off], tgt[off + NPLANE], tgt[off + 2 * NPLANE], l1, u1, v1);
    D[off]              = l0 - l1;
    D[off + NPLANE]     = u0 - u1;
    D[off + 2 * NPLANE] = v0 - v1;
}

// ---------------------------------------------------------------------------
// Shared filter body: given the 30x30 D tile already in LDS, do the
// horizontal 15-tap sliding sum and the vertical 15-tap pass via
// V_WMMA_F32_16X16X32_F16 (constant 0/1 band A; f16 hi+residual B split),
// then square-reduce to one partial per block.
// ---------------------------------------------------------------------------
__device__ __forceinline__ void filter_tail(
        float (&D)[3][30][32], _Float16 (&Yh)[3][16][32],
        _Float16 (&Yr)[3][16][32], float (&wsum)[3],
        int tid, int bx, float* __restrict__ partial) {
    // ---- horizontal 15-tap sliding sums -> Y[w][kk] (rows kk=1..30) ----
    if (tid < 90) {
        const int c = tid / 30, kk1 = tid % 30;           // kk = kk1 + 1
        float s = 0.f;
#pragma unroll
        for (int j = 0; j < 15; ++j) s += D[c][kk1][j];
#pragma unroll
        for (int w = 0; w < 16; ++w) {
            const _Float16 hi = (_Float16)s;
            Yh[c][w][kk1 + 1] = hi;
            Yr[c][w][kk1 + 1] = (_Float16)(s - (float)hi);
            if (w < 15) s += D[c][kk1][w + 15] - D[c][kk1][w];
        }
    }
    {   // zero kk=0 and kk=31 (exactly 96 border entries, one per thread)
        const int c = tid / 32, q = tid & 31;
        const int w = q >> 1, kk = (q & 1) * 31;
        Yh[c][w][kk] = (_Float16)0.f;
        Yr[c][w][kk] = (_Float16)0.f;
    }
    __syncthreads();

    // ---- vertical pass: O[m,w] = sum_kk band(m,kk) * Y[kk,w];
    // band ones for kk in [m+1, m+15] (window starts at 16*tr - 8).
    const int ch   = tid >> 5;
    const int lane = tid & 31;
    const int m    = lane & 15;
    const int hi   = lane >> 4;

    v16h a, bh, br;
#pragma unroll
    for (int e = 0; e < 16; ++e) {
        const int p = e >> 1;
        const int K = ((p & 4) << 2) + ((p & 3) << 1) + (hi << 3) + (e & 1);
        const int d = K - m;
        a[e]  = (_Float16)((d >= 1 && d <= 15) ? 1.0f : 0.0f);
        bh[e] = Yh[ch][m][K];
        br[e] = Yr[ch][m][K];
    }
    v8f acc = {};
    acc = __builtin_amdgcn_wmma_f32_16x16x32_f16(false, a, false, bh,
                                                 (short)0, acc, false, false);
    acc = __builtin_amdgcn_wmma_f32_16x16x32_f16(false, a, false, br,
                                                 (short)0, acc, false, false);

    float s = 0.f;
#pragma unroll
    for (int j = 0; j < 8; ++j) s += acc[j] * acc[j];
#pragma unroll
    for (int off = 16; off > 0; off >>= 1) s += __shfl_xor(s, off, 32);
    if (lane == 0) wsum[ch] = s;
    __syncthreads();
    if (tid == 0) partial[bx] = wsum[0] + wsum[1] + wsum[2];
}

// ---------------------------------------------------------------------------
// Kernel 2 (path A): load 30x30 D halo tile from global, filter + WMMA.
// ---------------------------------------------------------------------------
__global__ __launch_bounds__(96) void boxdiff_kernel(
        const float* __restrict__ D_g, float* __restrict__ partial) {
    __shared__ float    D[3][30][32];
    __shared__ _Float16 Yh[3][16][32];
    __shared__ _Float16 Yr[3][16][32];
    __shared__ float    wsum[3];

    const int bx = blockIdx.x;
    const int tc = bx & 31, tr = (bx >> 5) & 31, n = bx >> 10;
    const int tid = threadIdx.x;
    const int r0 = tr * 16 - 7, c0 = tc * 16 - 7;
    const size_t base = (size_t)n * NIMG;

    for (int p = tid; p < 900; p += 96) {
        const int row = p / 30, col = p % 30;
        const int gr = r0 + row, gc = c0 + col;
        float d0 = 0.f, d1 = 0.f, d2 = 0.f;
        if ((unsigned)gr < HW && (unsigned)gc < HW) {
            const size_t off = base + (size_t)gr * HW + gc;
            d0 = D_g[off];
            d1 = D_g[off + NPLANE];
            d2 = D_g[off + 2 * NPLANE];
        }
        D[0][row][col] = d0;
        D[1][row][col] = d1;
        D[2][row][col] = d2;
    }
    __syncthreads();
    filter_tail(D, Yh, Yr, wsum, tid, bx, partial);
}

// ---------------------------------------------------------------------------
// Path B fallback (tiny workspace): fully fused, converts the halo in-tile.
// ---------------------------------------------------------------------------
__global__ __launch_bounds__(96) void luv_boxdiff_fused_kernel(
        const float* __restrict__ inp, const float* __restrict__ tgt,
        float* __restrict__ partial) {
    __shared__ float    D[3][30][32];
    __shared__ _Float16 Yh[3][16][32];
    __shared__ _Float16 Yr[3][16][32];
    __shared__ float    wsum[3];

    const int bx = blockIdx.x;
    const int tc = bx & 31, tr = (bx >> 5) & 31, n = bx >> 10;
    const int tid = threadIdx.x;
    const int r0 = tr * 16 - 7, c0 = tc * 16 - 7;
    const size_t base = (size_t)n * NIMG;

    for (int p = tid; p < 900; p += 96) {
        const int row = p / 30, col = p % 30;
        const int gr = r0 + row, gc = c0 + col;
        float d0 = 0.f, d1 = 0.f, d2 = 0.f;
        if ((unsigned)gr < HW && (unsigned)gc < HW) {
            const size_t off = base + (size_t)gr * HW + gc;
            float l0, u0, v0, l1, u1, v1;
            rgb2luv(inp[off], inp[off + NPLANE], inp[off + 2 * NPLANE], l0, u0, v0);
            rgb2luv(tgt[off], tgt[off + NPLANE], tgt[off + 2 * NPLANE], l1, u1, v1);
            d0 = l0 - l1; d1 = u0 - u1; d2 = v0 - v1;
        }
        D[0][row][col] = d0;
        D[1][row][col] = d1;
        D[2][row][col] = d2;
    }
    __syncthreads();
    filter_tail(D, Yh, Yr, wsum, tid, bx, partial);
}

// ---------------------------------------------------------------------------
// Deterministic fixed-order final reduction (no float atomics).
// ---------------------------------------------------------------------------
__global__ __launch_bounds__(256) void reduce_kernel(
        const float* __restrict__ partial, float* __restrict__ out) {
    __shared__ float sm[256];
    const int tid = threadIdx.x;
    float s = 0.f;
    for (int i = tid; i < NTILES; i += 256) s += partial[i];
    sm[tid] = s;
    __syncthreads();
    for (int st = 128; st > 0; st >>= 1) {
        if (tid < st) sm[tid] += sm[tid + st];
        __syncthreads();
    }
    // mean over (N,H,W)=16*512*512 per channel, summed over channels
    if (tid == 0) out[0] = sm[0] * (1.0f / 4194304.0f);
}

extern "C" void kernel_launch(void* const* d_in, const int* in_sizes, int n_in,
                              void* d_out, int out_size, void* d_ws, size_t ws_size,
                              hipStream_t stream) {
    const float* inp = (const float*)d_in[0];
    const float* tgt = (const float*)d_in[1];
    float* out = (float*)d_out;

    const size_t D_BYTES = (size_t)NPIX * 3 * sizeof(float);   // 50,331,648
    const size_t NEEDED  = D_BYTES + NTILES * sizeof(float);

    if (ws_size >= NEEDED) {
        // Path A: convert once (no halo recompute), then filter from D.
        float* D_g     = (float*)d_ws;
        float* partial = (float*)((char*)d_ws + D_BYTES);
        luv_diff_kernel<<<NPIX / 256, 256, 0, stream>>>(inp, tgt, D_g);
        boxdiff_kernel<<<NTILES, 96, 0, stream>>>(D_g, partial);
        reduce_kernel<<<1, 256, 0, stream>>>(partial, out);
    } else {
        // Path B: fully fused fallback, needs only 64 KB of workspace.
        float* partial = (float*)d_ws;
        luv_boxdiff_fused_kernel<<<NTILES, 96, 0, stream>>>(inp, tgt, partial);
        reduce_kernel<<<1, 256, 0, stream>>>(partial, out);
    }
}
